// LSTMClassifier_51230369907035
// MI455X (gfx1250) — matrix operand order repaired and checked by reading, not compiled, not run on hardware
//
#include <hip/hip_runtime.h>
#include <hip/hip_bf16.h>

typedef __bf16 bf16;
typedef bf16  v16bf __attribute__((ext_vector_type(16)));
typedef bf16  v8bf  __attribute__((ext_vector_type(8)));
typedef float v8f   __attribute__((ext_vector_type(8)));

#define SCAN_NWG     64
#define SCAN_THREADS 256

#if __has_builtin(__builtin_amdgcn_global_load_async_to_lds_b128) && \
    __has_builtin(__builtin_amdgcn_s_wait_asynccnt)
#define USE_ASYNC_LDS 1
// Payload type expected by the async-LDS builtins on this toolchain:
// pointers to 128-bit int vectors in AS1 (global) / AS3 (LDS).
typedef int v4i __attribute__((vector_size(4 * sizeof(int))));
typedef __attribute__((address_space(1))) v4i gv4i_t;
typedef __attribute__((address_space(3))) v4i lv4i_t;
__device__ __forceinline__ gv4i_t* to_global(const void* p) {
    return (gv4i_t*)(unsigned long long)p;
}
__device__ __forceinline__ lv4i_t* to_lds(const void* p) {
    // Flat LDS aperture: low 32 address bits are the LDS address (ISA 10.2), so truncation is the offset.
    return (lv4i_t*)(unsigned int)(unsigned long long)p;
}
#else
#define USE_ASYNC_LDS 0
#endif

// ---------------------------------------------------------------------------
// Zero the grid-barrier counters (d_ws is poisoned to 0xAA by the harness).
// ---------------------------------------------------------------------------
__global__ void init_ws_kernel(unsigned* ctr) {
    if (threadIdx.x < 64) ctr[threadIdx.x] = 0u;
}

// ---------------------------------------------------------------------------
// Embedding gather with padding_idx=0 (row 0 forced to zero), fp32 -> bf16.
// ---------------------------------------------------------------------------
__global__ void embed_kernel(const int* __restrict__ seq,
                             const float* __restrict__ emb,
                             bf16* __restrict__ xbf, int E) {
    const int t = blockIdx.x;
    const int idx = seq[t];
    const float* src = emb + (size_t)idx * E;
    bf16* dst = xbf + (size_t)t * E;
    if (idx == 0) {
        for (int e = threadIdx.x; e < E; e += blockDim.x) dst[e] = (bf16)0.f;
    } else {
        for (int e = threadIdx.x; e < E; e += blockDim.x) dst[e] = (bf16)src[e];
    }
}

__global__ void f32_to_bf16_kernel(const float* __restrict__ src,
                                   bf16* __restrict__ dst, size_t n) {
    size_t i = (size_t)blockIdx.x * blockDim.x + threadIdx.x;
    const size_t stride = (size_t)gridDim.x * blockDim.x;
    for (; i < n; i += stride) dst[i] = (bf16)src[i];
}

// ---------------------------------------------------------------------------
// out[M,4096] = X[M,K](bf16) @ W[4096,K](bf16)^T + (bA + bB)  (fp32 accum)
// Each wave owns a 16(M) x 64(N) strip: the A fragment is loaded once per
// K-step and reused across 4 v_wmma_f32_16x16x32_bf16 (4 accumulators).
// Block = 8 waves -> 16 x 512; grid = (4096/512, M/16).
//
// WMMA lane layouts used here (A/B K order is this code's contiguous choice, same for A and B):
//   A 16x32 bf16: lane (h=lane>>4, m=lane&15); elem e -> K=e+16h
//   B 32x16 bf16: lane (h, n=lane&15);          elem e -> K=e+16h  (row n of W)
//   C/D 16x16 f32: vgpr r -> M=r+8h, N=lane&15
// ---------------------------------------------------------------------------
__global__ void __launch_bounds__(256)
gemm_bf16_wmma_kernel(const bf16* __restrict__ X,   // [M, K]
                      const bf16* __restrict__ W,   // [4096, K]
                      const float* __restrict__ bA,
                      const float* __restrict__ bB,
                      float* __restrict__ out,      // [M, 4096]
                      int K) {
    const int lane = threadIdx.x & 31;
    const int wave = threadIdx.x >> 5;
    const int h    = lane >> 4;
    const int ln   = lane & 15;
    const int n0   = blockIdx.x * 512 + wave * 64;
    const int m0   = blockIdx.y * 16;

    const bf16* xrow = X + (size_t)(m0 + ln) * K;
    const bf16* w0   = W + (size_t)(n0 +  0 + ln) * K;
    const bf16* w1   = W + (size_t)(n0 + 16 + ln) * K;
    const bf16* w2   = W + (size_t)(n0 + 32 + ln) * K;
    const bf16* w3   = W + (size_t)(n0 + 48 + ln) * K;

    v8f acc0 = {}, acc1 = {}, acc2 = {}, acc3 = {};
    for (int k0 = 0; k0 < K; k0 += 32) {
        if (k0 + 64 < K) {  // uniform branch; EXEC stays all-ones for WMMA
            __builtin_prefetch(xrow + k0 + 64, 0, 1);  // global_prefetch_b8
            __builtin_prefetch(w0 + k0 + 64, 0, 1);
            __builtin_prefetch(w1 + k0 + 64, 0, 1);
            __builtin_prefetch(w2 + k0 + 64, 0, 1);
            __builtin_prefetch(w3 + k0 + 64, 0, 1);
        }
        v8bf alo = *(const v8bf*)(xrow + k0 + 16 * h);
        v8bf ahi = *(const v8bf*)(xrow + k0 + 16 * h + 8);
        v16bf a = __builtin_shufflevector(alo, ahi, 0, 1, 2, 3, 4, 5, 6, 7,
                                          8, 9, 10, 11, 12, 13, 14, 15);
        v16bf b0 = *(const v16bf*)(w0 + k0 + 16 * h);
        v16bf b1 = *(const v16bf*)(w1 + k0 + 16 * h);
        v16bf b2 = *(const v16bf*)(w2 + k0 + 16 * h);
        v16bf b3 = *(const v16bf*)(w3 + k0 + 16 * h);
        acc0 = __builtin_amdgcn_wmma_f32_16x16x32_bf16(false, a, false, b0, (short)0, acc0, false, false);
        acc1 = __builtin_amdgcn_wmma_f32_16x16x32_bf16(false, a, false, b1, (short)0, acc1, false, false);
        acc2 = __builtin_amdgcn_wmma_f32_16x16x32_bf16(false, a, false, b2, (short)0, acc2, false, false);
        acc3 = __builtin_amdgcn_wmma_f32_16x16x32_bf16(false, a, false, b3, (short)0, acc3, false, false);
    }

#pragma unroll
    for (int nt = 0; nt < 4; ++nt) {
        const v8f acc = (nt == 0) ? acc0 : (nt == 1) ? acc1 : (nt == 2) ? acc2 : acc3;
        const int nc = n0 + nt * 16 + ln;
        const float bias = bA[nc] + bB[nc];
#pragma unroll
        for (int r = 0; r < 8; ++r)
            out[(size_t)(m0 + r + 8 * h) * 4096 + nc] = acc[r] + bias;
    }
}

// ---------------------------------------------------------------------------
// Persistent weight-stationary LSTM scan.
// 64 workgroups; WG owns 16 hidden dims -> 64 gate rows of W_hh (256 KB fp32)
// resident in LDS (CDNA5 WGP has 320 KB). Per step: stage h (4 KB) into LDS
// (async global->LDS when available), 64 dot-products (4 lanes/row, shfl_xor
// reduce), gate math for owned dims, double-buffered h in global, one
// device-scope barrier per step.
// ---------------------------------------------------------------------------
__device__ __forceinline__ void grid_barrier(unsigned* ctr, unsigned target) {
    __syncthreads();                 // all block stores precede arrival
    if (threadIdx.x == 0) {
        __threadfence();             // device-scope release of h writes
        __hip_atomic_fetch_add(ctr, 1u, __ATOMIC_ACQ_REL, __HIP_MEMORY_SCOPE_AGENT);
        while (__hip_atomic_load(ctr, __ATOMIC_ACQUIRE, __HIP_MEMORY_SCOPE_AGENT) < target)
            __builtin_amdgcn_s_sleep(1);
    }
    __syncthreads();
}

__global__ void __launch_bounds__(SCAN_THREADS)
lstm_scan_kernel(const float* __restrict__ pre,   // [T, 4096] gate pre-acts
                 const float* __restrict__ Whh,   // [4096, 1024]
                 float* __restrict__ h_seq,       // [T, 1024] out
                 float* __restrict__ hbuf,        // [2, 1024] double buffer
                 unsigned* __restrict__ ctr,
                 int T) {
    extern __shared__ float lds[];
    float* wl    = lds;               // 64*1024 weights (256 KB)
    float* hs    = wl + 64 * 1024;    // 1024 staged h
    float* gates = hs + 1024;         // 64
    float* cst   = gates + 64;        // 16 cell state

    const int tid = threadIdx.x;
    const int wg  = blockIdx.x;
    const int d0  = wg * 16;

    // Load this WG's 64 gate rows (gate g, dim d0+j -> W_hh row g*1024+d0+j).
    for (int idx = tid; idx < 64 * 1024; idx += SCAN_THREADS) {
        const int r = idx >> 10, k = idx & 1023;
        const int g = r >> 4, j = r & 15;
        wl[idx] = Whh[((size_t)(g * 1024 + d0 + j)) * 1024 + k];
    }
    if (tid < 16) {
        cst[tid] = 0.f;
        hbuf[d0 + tid] = 0.f;         // h(-1) = 0 in read buffer 0
    }

    unsigned tgt = SCAN_NWG;
    grid_barrier(ctr, tgt);           // h init visible everywhere

    const int row = tid >> 2;         // 0..63 gate row (g*16 + j)
    const int sub = tid & 3;          // quarter of the 1024-dot
    const float* wrow = &wl[row * 1024 + sub * 256];

    for (int t = 0; t < T; ++t) {
        const float* hread  = hbuf + ((t) & 1) * 1024;
        float*       hwrite = hbuf + ((t + 1) & 1) * 1024;

#if USE_ASYNC_LDS
        // 256 lanes x 16B = 4 KB: one async b128 per lane, LDS written directly.
        __builtin_amdgcn_global_load_async_to_lds_b128(
            to_global(hread + tid * 4), to_lds(hs + tid * 4), 0, 0);
        __builtin_amdgcn_s_wait_asynccnt(0);   // each wave drains its own ASYNCcnt
#else
        for (int i = tid; i < 1024; i += SCAN_THREADS) hs[i] = hread[i];
#endif
        __syncthreads();

        float acc = 0.f;
        const float* hv = hs + sub * 256;
#pragma unroll 8
        for (int k = 0; k < 256; ++k) acc = __builtin_fmaf(wrow[k], hv[k], acc);
        acc += __shfl_xor(acc, 1, 32);
        acc += __shfl_xor(acc, 2, 32);

        if (sub == 0) {
            const int g = row >> 4, j = row & 15;
            gates[row] = pre[(size_t)t * 4096 + g * 1024 + d0 + j] + acc;
        }
        __syncthreads();

        if (tid < 16) {
            const float iv = 1.f / (1.f + __expf(-gates[tid]));
            const float fv = 1.f / (1.f + __expf(-gates[16 + tid]));
            const float gv = tanhf(gates[32 + tid]);
            const float ov = 1.f / (1.f + __expf(-gates[48 + tid]));
            const float cc = fv * cst[tid] + iv * gv;
            cst[tid] = cc;
            const float hh = ov * tanhf(cc);
            hwrite[d0 + tid] = hh;
            h_seq[(size_t)t * 1024 + d0 + tid] = hh;
        }
        tgt += SCAN_NWG;
        grid_barrier(ctr, tgt);
    }
}

// ---------------------------------------------------------------------------
// logits = h2[T-1] @ W_out^T + b_out ; relu ; log_softmax  (C <= 16)
// ---------------------------------------------------------------------------
__global__ void __launch_bounds__(256)
head_kernel(const float* __restrict__ h2_seq, const float* __restrict__ Wout,
            const float* __restrict__ bout, float* __restrict__ out,
            int T, int C) {
    __shared__ float red[256];
    __shared__ float logits[16];
    const float* hlast = h2_seq + (size_t)(T - 1) * 1024;
    for (int c = 0; c < C; ++c) {
        float p = 0.f;
        for (int k = threadIdx.x; k < 1024; k += 256) p += hlast[k] * Wout[(size_t)c * 1024 + k];
        red[threadIdx.x] = p;
        __syncthreads();
        for (int s = 128; s > 0; s >>= 1) {
            if (threadIdx.x < s) red[threadIdx.x] += red[threadIdx.x + s];
            __syncthreads();
        }
        if (threadIdx.x == 0) logits[c] = red[0] + bout[c];
        __syncthreads();
    }
    if (threadIdx.x == 0) {
        float zh[16], m = 0.f;
        for (int c = 0; c < C; ++c) { zh[c] = fmaxf(logits[c], 0.f); m = fmaxf(m, zh[c]); }
        float s = 0.f;
        for (int c = 0; c < C; ++c) s += __expf(zh[c] - m);
        const float lse = logf(s) + m;
        for (int c = 0; c < C; ++c) out[c] = zh[c] - lse;
    }
}

// ---------------------------------------------------------------------------
extern "C" void kernel_launch(void* const* d_in, const int* in_sizes, int n_in,
                              void* d_out, int out_size, void* d_ws, size_t ws_size,
                              hipStream_t stream) {
    const int*   seq   = (const int*)d_in[0];
    const float* emb   = (const float*)d_in[1];
    const float* W_ih0 = (const float*)d_in[2];
    const float* W_hh0 = (const float*)d_in[3];
    const float* b_ih0 = (const float*)d_in[4];
    const float* b_hh0 = (const float*)d_in[5];
    const float* W_ih1 = (const float*)d_in[6];
    const float* W_hh1 = (const float*)d_in[7];
    const float* b_ih1 = (const float*)d_in[8];
    const float* b_hh1 = (const float*)d_in[9];
    const float* W_out = (const float*)d_in[10];
    const float* b_out = (const float*)d_in[11];
    float* out = (float*)d_out;

    const int T = in_sizes[0];            // 4096
    const int H = 1024;
    const int G = 4 * H;                  // 4096
    const int E = in_sizes[2] / G;        // 512
    const int C = in_sizes[11];           // 3

    // Workspace carve-out (256B aligned).
    char* ws = (char*)d_ws;
    size_t off = 0;
    auto carve = [&](size_t bytes) -> void* {
        void* p = ws + off;
        off = (off + bytes + 255) & ~(size_t)255;
        return p;
    };
    bf16*     xbf    = (bf16*)carve((size_t)T * E * 2);
    bf16*     wih0bf = (bf16*)carve((size_t)G * E * 2);
    bf16*     wih1bf = (bf16*)carve((size_t)G * H * 2);
    bf16*     h1bf   = (bf16*)carve((size_t)T * H * 2);
    float*    pre    = (float*)carve((size_t)T * G * 4);   // reused for both layers
    float*    h1seq  = (float*)carve((size_t)T * H * 4);
    float*    h2seq  = (float*)carve((size_t)T * H * 4);
    float*    hbuf   = (float*)carve((size_t)2 * H * 4);
    unsigned* ctr    = (unsigned*)carve(256);
    if (off > ws_size) return;  // workspace too small; deterministic no-op

    const size_t smem = (size_t)(64 * 1024 + 1024 + 64 + 16) * sizeof(float); // ~260 KB
    (void)hipFuncSetAttribute((const void*)lstm_scan_kernel,
                              hipFuncAttributeMaxDynamicSharedMemorySize, (int)smem);

    init_ws_kernel<<<1, 64, 0, stream>>>(ctr);
    embed_kernel<<<T, 256, 0, stream>>>(seq, emb, xbf, E);
    f32_to_bf16_kernel<<<512, 256, 0, stream>>>(W_ih0, wih0bf, (size_t)G * E);
    f32_to_bf16_kernel<<<512, 256, 0, stream>>>(W_ih1, wih1bf, (size_t)G * H);

    dim3 ggrid(G / 512, T / 16);
    // pre0 = x @ W_ih0^T + (b_ih0 + b_hh0)
    gemm_bf16_wmma_kernel<<<ggrid, 256, 0, stream>>>(xbf, wih0bf, b_ih0, b_hh0, pre, E);
    // layer-0 recurrence
    lstm_scan_kernel<<<SCAN_NWG, SCAN_THREADS, smem, stream>>>(pre, W_hh0, h1seq, hbuf, ctr, T);
    // pre1 = h1 @ W_ih1^T + (b_ih1 + b_hh1)
    f32_to_bf16_kernel<<<512, 256, 0, stream>>>(h1seq, h1bf, (size_t)T * H);
    gemm_bf16_wmma_kernel<<<ggrid, 256, 0, stream>>>(h1bf, wih1bf, b_ih1, b_hh1, pre, H);
    // layer-1 recurrence (fresh counter, read-buffer re-zeroed inside)
    lstm_scan_kernel<<<SCAN_NWG, SCAN_THREADS, smem, stream>>>(pre, W_hh1, h2seq, hbuf, ctr + 16, T);
    // classifier head
    head_kernel<<<1, 256, 0, stream>>>(h2seq, W_out, b_out, out, T, C);
}
